// SceneGraphGenerationModel_24507083391529
// MI455X (gfx1250) — matrix operand order repaired and checked by reading, not compiled, not run on hardware
//
#include <hip/hip_runtime.h>
#include <stdint.h>

// ---------------------------------------------------------------------------
// Types for CDNA5 WMMA
// ---------------------------------------------------------------------------
typedef __attribute__((ext_vector_type(16))) __bf16 v16bf;
typedef __attribute__((ext_vector_type(8)))  float  v8f;

__device__ __forceinline__ uint16_t f32_to_bf16(float f) {
  uint32_t u = __float_as_uint(f);
  uint32_t r = (u + 0x7FFFu + ((u >> 16) & 1u)) >> 16;  // round-to-nearest-even
  return (uint16_t)r;
}

// ---------------------------------------------------------------------------
// Constants (match reference)
// ---------------------------------------------------------------------------
#define BB    8
#define CC    256
#define HH    128
#define WW    128
#define NN    64
#define RR    7
#define EE    512
#define HID   256
#define NOBJ  150
#define NREL  50
#define NATTR 200
#define FEAT  (CC * RR * RR)          // 12544
#define NPAIR 4032                    // 64*63
#define MPAIR (BB * NPAIR)            // 32256

// ---------------------------------------------------------------------------
// Weight pad + transpose to bf16:  dst[n*Kp + k] = src[k*N + n]  (zero pad)
// ---------------------------------------------------------------------------
__global__ void padT_kernel(const float* __restrict__ src, uint16_t* __restrict__ dst,
                            int K, int N, int Kp, int Np) {
  int idx = blockIdx.x * 256 + threadIdx.x;
  if (idx >= Kp * Np) return;
  int n = idx / Kp;
  int k = idx - n * Kp;
  float v = (k < K && n < N) ? src[(size_t)k * N + n] : 0.0f;
  dst[idx] = f32_to_bf16(v);
}

// ---------------------------------------------------------------------------
// 2D integral image (cumsum along x then y) per (b,c) plane, via LDS tile
// ---------------------------------------------------------------------------
__global__ void integral_kernel(const float* __restrict__ feat, float* __restrict__ integ) {
  __shared__ float t[HH][WW + 1];          // +1 pad: conflict-free column pass
  int plane = blockIdx.x;                  // b*C + c
  const float* src = feat  + (size_t)plane * HH * WW;
  float*       dst = integ + (size_t)plane * HH * WW;
  int tx = threadIdx.x;                    // 0..127
  for (int r = 0; r < HH; ++r) t[r][tx] = src[r * WW + tx];
  __syncthreads();
  // row cumsum: thread tx owns row tx
  float s = 0.0f;
  for (int x = 0; x < WW; ++x) { s += t[tx][x]; t[tx][x] = s; }
  __syncthreads();
  // column cumsum: thread tx owns column tx
  s = 0.0f;
  for (int y = 0; y < HH; ++y) { s += t[y][tx]; t[y][tx] = s; }
  __syncthreads();
  for (int r = 0; r < HH; ++r) dst[r * WW + tx] = t[r][tx];
}

// Padded integral lookup: I(y,x), y,x in [0,128]; 0 on the padded edge
__device__ __forceinline__ float ilook(const float* p, int y, int x) {
  return (y > 0 && x > 0) ? p[(y - 1) * WW + (x - 1)] : 0.0f;
}

// ---------------------------------------------------------------------------
// ROI pooling -> bf16 A-matrix (512 rows x 12544 cols), one block per box
// ---------------------------------------------------------------------------
__global__ void roi_kernel(const float* __restrict__ integ,
                           const float* __restrict__ boxes,
                           uint16_t* __restrict__ roiB) {
  int bn = blockIdx.x;                 // b*64 + n
  int b  = bn >> 6;
  const float* bx = boxes + (size_t)bn * 4;
  float cx = bx[0], cy = bx[1], w = bx[2], h = bx[3];
  int x1 = (int)fminf(fmaxf((cx - w * 0.5f) * (float)WW, 0.0f), (float)(WW - 1));
  int y1 = (int)fminf(fmaxf((cy - h * 0.5f) * (float)HH, 0.0f), (float)(HH - 1));
  int x2 = (int)fminf(fmaxf((cx + w * 0.5f) * (float)WW, 0.0f), (float)(WW - 1));
  int y2 = (int)fminf(fmaxf((cy + h * 0.5f) * (float)HH, 0.0f), (float)(HH - 1));
  int hb = y2 - y1, wb = x2 - x1;
  bool valid = (hb > 0) && (wb > 0);
  for (int idx = threadIdx.x; idx < FEAT; idx += 256) {
    int c   = idx / (RR * RR);
    int bin = idx - c * (RR * RR);
    int ry  = bin / RR;
    int rx  = bin - ry * RR;
    const float* p = integ + ((size_t)b * CC + c) * (HH * WW);
    int ys = y1 + (ry * hb) / RR;
    int ye = y1 + ((ry + 1) * hb + RR - 1) / RR;
    int xs = x1 + (rx * wb) / RR;
    int xe = x1 + ((rx + 1) * wb + RR - 1) / RR;
    float sum  = ilook(p, ye, xe) - ilook(p, ys, xe) - ilook(p, ye, xs) + ilook(p, ys, xs);
    float area = (float)((ye - ys) * (xe - xs));
    float v = sum / fmaxf(area, 1.0f);
    if (!valid) v = 0.0f;
    roiB[(size_t)bn * FEAT + idx] = f32_to_bf16(v);
  }
}

// ---------------------------------------------------------------------------
// Register-blocked WMMA GEMM:
//   out = act(A[M,Kp](bf16) @ Wt[Np,Kp]^T(bf16) + bias)
// One wave owns a BM x BN grid of 16x16 tiles (e.g. 64x64 output for 4x4):
// per K-step it loads BM A-fragments + BN B-fragments and issues BM*BN
// v_wmma_f32_16x16x32_bf16, giving BNx / BMx register reuse of A / B.
// Fragment layouts per CDNA5 ISA 7.12.2:
//   A (16x32, 16b): lane&15 = M; vgpr i holds K pair at (i<4?0:16)+(lane>>4)*8+(i&3)*2
//   B (32x16, 16b): lane&15 = N; vgpr i holds K pair at (i<4?0:16)+(lane>>4)*8+(i&3)*2
//   C/D (16x16 f32): vgpr j -> M = j + (lane>>4)*8, N = lane&15
// ---------------------------------------------------------------------------
template <int BM, int BN>
__global__ void __launch_bounds__(32)
gemm_wmma_blocked(const uint16_t* __restrict__ A, const uint16_t* __restrict__ Wt,
                  const float* __restrict__ bias, float* __restrict__ outF,
                  uint16_t* __restrict__ outB,
                  int Kp, int Np, int Nreal, int doRelu) {
  int lane = threadIdx.x;
  int ml = lane & 15, hl = lane >> 4;
  int hl4 = hl * 4, hl8 = hl * 8;
  int tm0 = blockIdx.y * BM;
  int tn0 = blockIdx.x * BN;
  const uint32_t* Abase = (const uint32_t*)(A  + (size_t)(tm0 * 16 + ml) * Kp);
  const uint32_t* Bbase = (const uint32_t*)(Wt + (size_t)(tn0 * 16 + ml) * Kp);
  const int strideT = 8 * Kp;               // dwords between consecutive 16-row tiles

  const v8f vzero = {0.f, 0.f, 0.f, 0.f, 0.f, 0.f, 0.f, 0.f};
  v8f acc[BM][BN];
#pragma unroll
  for (int im = 0; im < BM; ++im)
#pragma unroll
    for (int in = 0; in < BN; ++in) acc[im][in] = vzero;

  for (int k0 = 0; k0 < Kp; k0 += 32) {
    int kd = k0 >> 1;                       // dword index (2 bf16 per dword)
    union Frag { v16bf v; uint32_t u[8]; };
    Frag af[BM], bf[BN];
#pragma unroll
    for (int im = 0; im < BM; ++im) {
      const uint32_t* Ar = Abase + (size_t)im * strideT + kd;
      __builtin_prefetch(Ar + 128, 0, 1);   // 512B ahead on the A stream
#pragma unroll
      for (int i = 0; i < 8; ++i)
        af[im].u[i] = Ar[hl4 + i + (i >= 4 ? 4 : 0)];
    }
#pragma unroll
    for (int in = 0; in < BN; ++in) {
      const uint32_t* Br = Bbase + (size_t)in * strideT + kd;
#pragma unroll
      for (int i = 0; i < 8; ++i)
        bf[in].u[i] = Br[hl4 + i + (i >= 4 ? 4 : 0)];
    }
#pragma unroll
    for (int im = 0; im < BM; ++im)
#pragma unroll
      for (int in = 0; in < BN; ++in)
        acc[im][in] = __builtin_amdgcn_wmma_f32_16x16x32_bf16(
            /*neg_a=*/false, af[im].v, /*neg_b=*/false, bf[in].v,
            /*c_mod=*/(short)0, acc[im][in], /*reuse_a=*/false, /*reuse_b=*/false);
  }

#pragma unroll
  for (int im = 0; im < BM; ++im) {
#pragma unroll
    for (int in = 0; in < BN; ++in) {
      int col = (tn0 + in) * 16 + ml;
#pragma unroll
      for (int j = 0; j < 8; ++j) {
        int row = (tm0 + im) * 16 + hl8 + j;
        if (col < Nreal) {
          float v = acc[im][in][j] + bias[col];
          if (doRelu) v = fmaxf(v, 0.0f);
          if (outF) outF[(size_t)row * Nreal + col] = v;
          if (outB) outB[(size_t)row * Np + col] = f32_to_bf16(v);
        } else if (outB) {
          outB[(size_t)row * Np + col] = (uint16_t)0;  // keep padded cols zero
        }
      }
    }
  }
}

// ---------------------------------------------------------------------------
// Class-embedding gather -> bf16 (512 rows x 512 cols)
// ---------------------------------------------------------------------------
__global__ void emb_kernel(const float* __restrict__ tbl, const int* __restrict__ cls,
                           uint16_t* __restrict__ out) {
  int idx = blockIdx.x * 256 + threadIdx.x;      // < 512*512
  int row = idx >> 9;
  int e   = idx & 511;
  int c   = cls[row];
  out[idx] = f32_to_bf16(tbl[(size_t)c * EE + e]);
}

// ---------------------------------------------------------------------------
// Pair spatial features (10 real dims, padded to 32) -> bf16
// ---------------------------------------------------------------------------
__global__ void spin_kernel(const float* __restrict__ boxes, uint16_t* __restrict__ out) {
  int row = blockIdx.x * 256 + threadIdx.x;
  if (row >= MPAIR) return;
  int b = row / NPAIR;
  int p = row - b * NPAIR;
  int s = p / 63;
  int r = p - s * 63;
  int o = (r < s) ? r : r + 1;
  const float* sb = boxes + ((size_t)(b * NN + s)) * 4;
  const float* ob = boxes + ((size_t)(b * NN + o)) * 4;
  float v[10];
  v[0] = sb[0]; v[1] = sb[1]; v[2] = sb[2]; v[3] = sb[3];
  v[4] = ob[0]; v[5] = ob[1]; v[6] = ob[2]; v[7] = ob[3];
  v[8] = sb[0] - ob[0]; v[9] = sb[1] - ob[1];
  uint16_t* dst = out + (size_t)row * 32;
#pragma unroll
  for (int i = 0; i < 10; ++i) dst[i] = f32_to_bf16(v[i]);
#pragma unroll
  for (int i = 10; i < 32; ++i) dst[i] = 0;
}

// ---------------------------------------------------------------------------
// Build f = [subj_emb(512) | obj_emb(512) | sp(128)]  (bf16 copies)
// ---------------------------------------------------------------------------
__global__ void fbuild_kernel(const uint16_t* __restrict__ emb,
                              const uint16_t* __restrict__ sp,
                              uint16_t* __restrict__ f) {
  size_t idx = (size_t)blockIdx.x * 256 + threadIdx.x;
  const size_t total = (size_t)MPAIR * 1152;
  if (idx >= total) return;
  int row = (int)(idx / 1152);
  int col = (int)(idx - (size_t)row * 1152);
  int b = row / NPAIR;
  int p = row - b * NPAIR;
  int s = p / 63;
  int r = p - s * 63;
  int o = (r < s) ? r : r + 1;
  uint16_t v;
  if (col < 512)       v = emb[((size_t)(b * NN + s)) * EE + col];
  else if (col < 1024) v = emb[((size_t)(b * NN + o)) * EE + (col - 512)];
  else                 v = sp[(size_t)row * 128 + (col - 1024)];
  f[idx] = v;
}

// ---------------------------------------------------------------------------
// Host launcher
// ---------------------------------------------------------------------------
extern "C" void kernel_launch(void* const* d_in, const int* in_sizes, int n_in,
                              void* d_out, int out_size, void* d_ws, size_t ws_size,
                              hipStream_t stream) {
  (void)in_sizes; (void)n_in; (void)out_size; (void)ws_size;

  const float* features = (const float*)d_in[0];
  const float* boxes    = (const float*)d_in[1];
  const int*   classes  = (const int*)  d_in[2];
  const float* w_emb    = (const float*)d_in[3];
  const float* b_emb    = (const float*)d_in[4];
  const float* w_obj    = (const float*)d_in[5];
  const float* b_obj    = (const float*)d_in[6];
  const float* w_attr   = (const float*)d_in[7];
  const float* b_attr   = (const float*)d_in[8];
  const float* w_bbox   = (const float*)d_in[9];
  const float* b_bbox   = (const float*)d_in[10];
  const float* cls_tbl  = (const float*)d_in[11];
  const float* w_s1     = (const float*)d_in[12];
  const float* b_s1     = (const float*)d_in[13];
  const float* w_s2     = (const float*)d_in[14];
  const float* b_s2     = (const float*)d_in[15];
  const float* w_f1     = (const float*)d_in[16];
  const float* b_f1     = (const float*)d_in[17];
  const float* w_f2     = (const float*)d_in[18];
  const float* b_f2     = (const float*)d_in[19];
  const float* w_rel    = (const float*)d_in[20];
  const float* b_rel    = (const float*)d_in[21];
  float* out = (float*)d_out;

  // ---- workspace carving (256B aligned) ----
  char* ws = (char*)d_ws;
  size_t off = 0;
  auto alloc = [&](size_t bytes) -> void* {
    void* p = ws + off;
    off = (off + bytes + 255) & ~(size_t)255;
    return p;
  };
  float*    integ    = (float*)   alloc((size_t)BB * CC * HH * WW * 4);   // 134 MB
  uint16_t* roiB     = (uint16_t*)alloc((size_t)BB * NN * FEAT * 2);      // 12.8 MB
  uint16_t* w_emb_t  = (uint16_t*)alloc((size_t)EE * FEAT * 2);           // 12.8 MB
  uint16_t* objfB    = (uint16_t*)alloc((size_t)BB * NN * EE * 2);
  uint16_t* w_obj_t  = (uint16_t*)alloc((size_t)160 * EE * 2);
  uint16_t* w_attr_t = (uint16_t*)alloc((size_t)208 * EE * 2);
  uint16_t* w_bbox_t = (uint16_t*)alloc((size_t)16 * EE * 2);
  uint16_t* embB     = (uint16_t*)alloc((size_t)BB * NN * EE * 2);
  uint16_t* w_s1_t   = (uint16_t*)alloc((size_t)64 * 32 * 2);
  uint16_t* w_s2_t   = (uint16_t*)alloc((size_t)128 * 64 * 2);
  uint16_t* spinB    = (uint16_t*)alloc((size_t)MPAIR * 32 * 2);
  uint16_t* s1B      = (uint16_t*)alloc((size_t)MPAIR * 64 * 2);
  uint16_t* s2B      = (uint16_t*)alloc((size_t)MPAIR * 128 * 2);
  uint16_t* fB       = (uint16_t*)alloc((size_t)MPAIR * 1152 * 2);        // 74 MB
  uint16_t* w_f1_t   = (uint16_t*)alloc((size_t)HID * 1152 * 2);
  uint16_t* h1B      = (uint16_t*)alloc((size_t)MPAIR * HID * 2);
  uint16_t* w_f2_t   = (uint16_t*)alloc((size_t)HID * HID * 2);
  uint16_t* h2B      = (uint16_t*)alloc((size_t)MPAIR * HID * 2);
  uint16_t* w_rel_t  = (uint16_t*)alloc((size_t)64 * HID * 2);

  auto padT = [&](const float* src, uint16_t* dst, int K, int N, int Kp, int Np) {
    int total = Kp * Np;
    padT_kernel<<<(total + 255) / 256, 256, 0, stream>>>(src, dst, K, N, Kp, Np);
  };

  // ---- weights -> bf16, transposed (N x K) with zero padding ----
  padT(w_emb,  w_emb_t,  FEAT, EE,   FEAT, EE);
  padT(w_obj,  w_obj_t,  EE,   NOBJ, EE,   160);
  padT(w_attr, w_attr_t, EE,   NATTR,EE,   208);
  padT(w_bbox, w_bbox_t, EE,   4,    EE,   16);
  padT(w_s1,   w_s1_t,   10,   64,   32,   64);
  padT(w_s2,   w_s2_t,   64,   128,  64,   128);
  padT(w_f1,   w_f1_t,   1152, HID,  1152, HID);
  padT(w_f2,   w_f2_t,   HID,  HID,  HID,  HID);
  padT(w_rel,  w_rel_t,  HID,  NREL, HID,  64);

  // ---- integral image + ROI pool ----
  integral_kernel<<<BB * CC, 128, 0, stream>>>(features, integ);
  roi_kernel<<<BB * NN, 256, 0, stream>>>(integ, boxes, roiB);

  // ---- obj_feats = relu(roi @ w_emb + b_emb)  [512 x 512] ----
  gemm_wmma_blocked<4, 4><<<dim3(EE / 64, (BB * NN) / 64), 32, 0, stream>>>(
      roiB, w_emb_t, b_emb, nullptr, objfB, FEAT, EE, EE, 1);

  // ---- heads: obj / attr / bbox logits (f32 into d_out) ----
  float* out_obj  = out;                       // 8*64*150
  float* out_attr = out + 76800;               // 8*64*200
  float* out_bbox = out + 179200;              // 8*64*4
  float* out_rel  = out + 181248;              // 8*4032*50
  gemm_wmma_blocked<4, 2><<<dim3(160 / 32, (BB * NN) / 64), 32, 0, stream>>>(
      objfB, w_obj_t, b_obj, out_obj, nullptr, EE, 160, NOBJ, 0);
  gemm_wmma_blocked<4, 1><<<dim3(208 / 16, (BB * NN) / 64), 32, 0, stream>>>(
      objfB, w_attr_t, b_attr, out_attr, nullptr, EE, 208, NATTR, 0);
  gemm_wmma_blocked<4, 1><<<dim3(16 / 16, (BB * NN) / 64), 32, 0, stream>>>(
      objfB, w_bbox_t, b_bbox, out_bbox, nullptr, EE, 16, 4, 0);

  // ---- pair pipeline ----
  emb_kernel<<<(BB * NN * EE) / 256, 256, 0, stream>>>(cls_tbl, classes, embB);
  spin_kernel<<<(MPAIR + 255) / 256, 256, 0, stream>>>(boxes, spinB);
  gemm_wmma_blocked<4, 4><<<dim3(64 / 64, MPAIR / 64), 32, 0, stream>>>(
      spinB, w_s1_t, b_s1, nullptr, s1B, 32, 64, 64, 1);
  gemm_wmma_blocked<4, 4><<<dim3(128 / 64, MPAIR / 64), 32, 0, stream>>>(
      s1B, w_s2_t, b_s2, nullptr, s2B, 64, 128, 128, 1);

  size_t ftotal = (size_t)MPAIR * 1152;
  fbuild_kernel<<<(unsigned)((ftotal + 255) / 256), 256, 0, stream>>>(embB, s2B, fB);

  gemm_wmma_blocked<4, 4><<<dim3(HID / 64, MPAIR / 64), 32, 0, stream>>>(
      fB, w_f1_t, b_f1, nullptr, h1B, MPAIR == 0 ? 0 : 1152, HID, HID, 1);
  gemm_wmma_blocked<4, 4><<<dim3(HID / 64, MPAIR / 64), 32, 0, stream>>>(
      h1B, w_f2_t, b_f2, nullptr, h2B, HID, HID, HID, 1);
  gemm_wmma_blocked<4, 4><<<dim3(64 / 64, MPAIR / 64), 32, 0, stream>>>(
      h2B, w_rel_t, b_rel, out_rel, nullptr, HID, 64, NREL, 0);
}
